// HierarchicalCapsuleNetwork_16578573763143
// MI455X (gfx1250) — matrix operand order repaired and checked by reading, not compiled, not run on hardware
//
#include <hip/hip_runtime.h>
#include <hip/hip_bf16.h>

typedef __attribute__((ext_vector_type(16))) _Float16 v16h;
typedef __attribute__((ext_vector_type(8)))  _Float16 v8h;
typedef __attribute__((ext_vector_type(8)))  float    v8f;

#define LDSTR 40   // padded LDS row stride in halves (80B: 16B-aligned, staggers banks)

// ---------------------------------------------------------------------------
// Prep 1: zero-pad features and convert to f16: (64,512,28,28)f32 -> (64,512,30,30)f16
// ---------------------------------------------------------------------------
__global__ __launch_bounds__(256) void pad_feat_kernel(
    const float* __restrict__ feat, _Float16* __restrict__ xpad)
{
    int idx = blockIdx.x * 256 + threadIdx.x;          // < 29,491,200
    int bc = idx / 900;
    int hw = idx - bc * 900;
    int h  = hw / 30;
    int w  = hw - h * 30;
    float v = 0.f;
    if (h >= 1 && h <= 28 && w >= 1 && w <= 28)
        v = feat[(size_t)bc * 784 + (h - 1) * 28 + (w - 1)];
    xpad[idx] = (_Float16)v;
}

// ---------------------------------------------------------------------------
// Prep 2: weights -> f16 in K' order, k' = (kh*3+kw)*512 + ci   (conv1)
// ---------------------------------------------------------------------------
__global__ __launch_bounds__(256) void w1_transform_kernel(
    const float* __restrict__ w, _Float16* __restrict__ out)
{
    int idx = blockIdx.x * 256 + threadIdx.x;          // < 1,179,648
    int n    = idx / 4608;
    int kp   = idx - n * 4608;
    int khkw = kp >> 9;
    int ci   = kp & 511;
    out[idx] = (_Float16)w[(size_t)n * 4608 + ci * 9 + khkw];
}

// ---------------------------------------------------------------------------
// Prep 3: weights -> f16 in K' order, k' = (kh*9+kw)*256 + ci   (conv2)
// ---------------------------------------------------------------------------
__global__ __launch_bounds__(256) void w2_transform_kernel(
    const float* __restrict__ w, _Float16* __restrict__ out)
{
    int idx = blockIdx.x * 256 + threadIdx.x;          // < 5,308,416
    int n    = idx / 20736;
    int kp   = idx - n * 20736;
    int khkw = kp >> 8;
    int ci   = kp & 255;
    out[idx] = (_Float16)w[(size_t)n * 20736 + ci * 81 + khkw];
}

// ---------------------------------------------------------------------------
// Kernel 1: conv1 (512->256, k3, pad1) + bias + BN(eval) + ReLU, implicit GEMM
// M = 50176, N = 256 (full N per WG), K = 4608 reordered as 9 x 512.
// 256 threads = 8 waves in a 2(M) x 4(N) grid; 64x256 tile; ping-pong LDS.
// ---------------------------------------------------------------------------
__global__ __launch_bounds__(256) void conv1_bn_relu_kernel(
    const _Float16* __restrict__ xpad, const _Float16* __restrict__ w1t,
    const float* __restrict__ bias, const float* __restrict__ gamma,
    const float* __restrict__ beta, const float* __restrict__ mean,
    const float* __restrict__ var, _Float16* __restrict__ xout)
{
    __shared__ __align__(16) _Float16 As[2][64 * LDSTR];
    __shared__ __align__(16) _Float16 Bs[2][256 * LDSTR];

    const int tid    = threadIdx.x;
    const int blockM = blockIdx.x;          // 0..783

    // A staging: 64 rows x 32 k, 4 threads per row (8 halves each)
    const int arow  = tid >> 2;
    const int kseg8 = (tid & 3) * 8;
    const int m     = blockM * 64 + arow;
    const int bimg  = m / 784;
    const int rem   = m - bimg * 784;
    const int ho    = rem / 28;
    const int wo    = rem - ho * 28;
    const size_t batchoff = (size_t)bimg * 512 * 900;
    // B staging: one weight row per thread
    const _Float16* __restrict__ wrow = w1t + (size_t)tid * 4608;

    const int wv   = tid >> 5;
    const int lane = tid & 31;
    const int g    = lane >> 4;
    const int lr   = lane & 15;
    const int waveM = (wv >> 2) * 32;       // 0 or 32
    const int waveN = (wv & 3) * 64;        // 0,64,128,192

    v8f acc[2][4] = {};
    _Float16 ta[8];
    v8h tb[4];

    auto load_tiles = [&](int it) {
        int khkw = it >> 4;                 // uniform scalar
        int cb   = (it & 15) << 5;
        int kh   = khkw / 3;
        int kw   = khkw - kh * 3;
        int pix  = (ho + kh) * 30 + (wo + kw);
        const _Float16* __restrict__ ap = xpad + batchoff + (size_t)(cb + kseg8) * 900 + pix;
        #pragma unroll
        for (int j = 0; j < 8; ++j) ta[j] = ap[j * 900];    // const-offset batched loads
        const _Float16* __restrict__ bp = wrow + khkw * 512 + cb;
        #pragma unroll
        for (int q = 0; q < 4; ++q) tb[q] = *(const v8h*)(bp + 8 * q);
    };

    load_tiles(0);
    *(v8h*)(As[0] + arow * LDSTR + kseg8) = *(v8h*)&ta[0];
    #pragma unroll
    for (int q = 0; q < 4; ++q) *(v8h*)(Bs[0] + tid * LDSTR + 8 * q) = tb[q];
    __syncthreads();

    for (int it = 0; it < 144; ++it) {
        const int pb = it & 1;
        if (it + 1 < 144) load_tiles(it + 1);   // overlap next gather with WMMAs

        v16h afr[2];
        #pragma unroll
        for (int tm = 0; tm < 2; ++tm) {
            int rowa = waveM + tm * 16 + lr;
            ((v8h*)&afr[tm])[0] = *(v8h*)(As[pb] + rowa * LDSTR + 8 * g);
            ((v8h*)&afr[tm])[1] = *(v8h*)(As[pb] + rowa * LDSTR + 16 + 8 * g);
        }
        #pragma unroll
        for (int tn = 0; tn < 4; ++tn) {
            int rowb = waveN + tn * 16 + lr;
            v16h b;
            ((v8h*)&b)[0] = *(v8h*)(Bs[pb] + rowb * LDSTR + 8 * g);
            ((v8h*)&b)[1] = *(v8h*)(Bs[pb] + rowb * LDSTR + 16 + 8 * g);
            #pragma unroll
            for (int tm = 0; tm < 2; ++tm)
                acc[tm][tn] = __builtin_amdgcn_wmma_f32_16x16x32_f16(
                    false, afr[tm], false, b, (short)0, acc[tm][tn], false, false);
        }
        if (it + 1 < 144) {                 // stage next tile into the idle buffer
            const int qb = 1 - pb;
            *(v8h*)(As[qb] + arow * LDSTR + kseg8) = *(v8h*)&ta[0];
            #pragma unroll
            for (int q = 0; q < 4; ++q) *(v8h*)(Bs[qb] + tid * LDSTR + 8 * q) = tb[q];
        }
        __syncthreads();                    // single barrier per K-step
    }

    // epilogue: bias + BN + ReLU, write f16 NCHW
    #pragma unroll
    for (int tn = 0; tn < 4; ++tn) {
        int co    = waveN + tn * 16 + lr;
        float bi  = bias[co];
        float inv = gamma[co] * rsqrtf(var[co] + 1e-5f);
        float sh  = beta[co] - mean[co] * inv;
        #pragma unroll
        for (int tm = 0; tm < 2; ++tm) {
            #pragma unroll
            for (int r = 0; r < 8; ++r) {
                int mm = blockM * 64 + waveM + tm * 16 + 8 * g + r;
                int bb = mm / 784;
                int rr = mm - bb * 784;
                int hh = rr / 28;
                int ww = rr - hh * 28;
                float v = acc[tm][tn][r] + bi;
                v = fmaxf(v * inv + sh, 0.f);
                xout[(((size_t)bb * 256 + co) * 28 + hh) * 28 + ww] = (_Float16)v;
            }
        }
    }
}

// ---------------------------------------------------------------------------
// Kernel 2: primary caps conv (256->256, k9, s2, valid) + bias, implicit GEMM
// M = 6400, N = 256 (full N per WG), K = 20736 reordered as 81 x 256.
// Output p[b][route][dim] f32.
// ---------------------------------------------------------------------------
__global__ __launch_bounds__(256) void conv2_prim_kernel(
    const _Float16* __restrict__ xin, const _Float16* __restrict__ w2t,
    const float* __restrict__ bias, float* __restrict__ p)
{
    __shared__ __align__(16) _Float16 As[2][64 * LDSTR];
    __shared__ __align__(16) _Float16 Bs[2][256 * LDSTR];

    const int tid    = threadIdx.x;
    const int blockM = blockIdx.x;          // 0..99

    const int arow  = tid >> 2;
    const int kseg8 = (tid & 3) * 8;
    const int m     = blockM * 64 + arow;
    const int bimg  = m / 100;
    const int s0    = m - bimg * 100;
    const int ho    = s0 / 10;
    const int wo    = s0 - ho * 10;
    const size_t batchoff = (size_t)bimg * 256 * 784;
    const _Float16* __restrict__ wrow = w2t + (size_t)tid * 20736;

    const int wv   = tid >> 5;
    const int lane = tid & 31;
    const int g    = lane >> 4;
    const int lr   = lane & 15;
    const int waveM = (wv >> 2) * 32;
    const int waveN = (wv & 3) * 64;

    v8f acc[2][4] = {};
    _Float16 ta[8];
    v8h tb[4];

    auto load_tiles = [&](int it) {
        int khkw = it >> 3;                 // uniform scalar, 0..80
        int cb   = (it & 7) << 5;
        int kh   = khkw / 9;
        int kw   = khkw - kh * 9;
        int pix  = (2 * ho + kh) * 28 + (2 * wo + kw);   // always in-bounds
        const _Float16* __restrict__ ap = xin + batchoff + (size_t)(cb + kseg8) * 784 + pix;
        #pragma unroll
        for (int j = 0; j < 8; ++j) ta[j] = ap[j * 784];
        const _Float16* __restrict__ bp = wrow + khkw * 256 + cb;
        #pragma unroll
        for (int q = 0; q < 4; ++q) tb[q] = *(const v8h*)(bp + 8 * q);
    };

    load_tiles(0);
    *(v8h*)(As[0] + arow * LDSTR + kseg8) = *(v8h*)&ta[0];
    #pragma unroll
    for (int q = 0; q < 4; ++q) *(v8h*)(Bs[0] + tid * LDSTR + 8 * q) = tb[q];
    __syncthreads();

    for (int it = 0; it < 648; ++it) {
        const int pb = it & 1;
        if (it + 1 < 648) load_tiles(it + 1);

        v16h afr[2];
        #pragma unroll
        for (int tm = 0; tm < 2; ++tm) {
            int rowa = waveM + tm * 16 + lr;
            ((v8h*)&afr[tm])[0] = *(v8h*)(As[pb] + rowa * LDSTR + 8 * g);
            ((v8h*)&afr[tm])[1] = *(v8h*)(As[pb] + rowa * LDSTR + 16 + 8 * g);
        }
        #pragma unroll
        for (int tn = 0; tn < 4; ++tn) {
            int rowb = waveN + tn * 16 + lr;
            v16h b;
            ((v8h*)&b)[0] = *(v8h*)(Bs[pb] + rowb * LDSTR + 8 * g);
            ((v8h*)&b)[1] = *(v8h*)(Bs[pb] + rowb * LDSTR + 16 + 8 * g);
            #pragma unroll
            for (int tm = 0; tm < 2; ++tm)
                acc[tm][tn] = __builtin_amdgcn_wmma_f32_16x16x32_f16(
                    false, afr[tm], false, b, (short)0, acc[tm][tn], false, false);
        }
        if (it + 1 < 648) {
            const int qb = 1 - pb;
            *(v8h*)(As[qb] + arow * LDSTR + kseg8) = *(v8h*)&ta[0];
            #pragma unroll
            for (int q = 0; q < 4; ++q) *(v8h*)(Bs[qb] + tid * LDSTR + 8 * q) = tb[q];
        }
        __syncthreads();
    }

    // epilogue: bias, scatter into capsule layout p[b][cap*100+s][dim]
    #pragma unroll
    for (int tn = 0; tn < 4; ++tn) {
        int co   = waveN + tn * 16 + lr;
        int cap  = co >> 5;
        int dim  = co & 31;
        float bi = bias[co];
        #pragma unroll
        for (int tm = 0; tm < 2; ++tm) {
            #pragma unroll
            for (int r = 0; r < 8; ++r) {
                int mm = blockM * 64 + waveM + tm * 16 + 8 * g + r;
                int bb = mm / 100;
                int ss = mm - bb * 100;
                int route = cap * 100 + ss;
                p[((size_t)bb * 800 + route) * 32 + dim] = acc[tm][tn][r] + bi;
            }
        }
    }
}

// ---------------------------------------------------------------------------
// Kernel 3: squash lower caps. One wave per 32-dim vector (lane == dim).
// ---------------------------------------------------------------------------
__global__ __launch_bounds__(256) void squash_kernel(
    const float* __restrict__ p, float* __restrict__ lower)
{
    int vec  = blockIdx.x * 8 + (threadIdx.x >> 5);
    int lane = threadIdx.x & 31;
    float x = p[(size_t)vec * 32 + lane];
    float n2 = x * x;
    n2 += __shfl_xor(n2, 16, 32);
    n2 += __shfl_xor(n2, 8, 32);
    n2 += __shfl_xor(n2, 4, 32);
    n2 += __shfl_xor(n2, 2, 32);
    n2 += __shfl_xor(n2, 1, 32);
    float scale = (n2 / (1.f + n2)) / sqrtf(n2 + 1e-8f);
    lower[(size_t)vec * 32 + lane] = x * scale;
}

// ---------------------------------------------------------------------------
// Kernel 4: u_hat = einsum(bri, rjoi -> brjo). One WG per route r:
// GEMM M=64(batch) x N=32(j*16+o) x K=32(i) with WMMA.
// ---------------------------------------------------------------------------
__global__ __launch_bounds__(128) void uhat_kernel(
    const float* __restrict__ lower, const float* __restrict__ rw,
    float* __restrict__ uhat)
{
    __shared__ __align__(16) _Float16 As[64 * LDSTR];
    __shared__ __align__(16) _Float16 Bs[32 * LDSTR];

    const int r   = blockIdx.x;             // 0..799
    const int tid = threadIdx.x;

    {   // A: lower[b][r][i], 64 rows x 32 k
        int row  = tid >> 1;
        int kseg = (tid & 1) * 16;
        const float* __restrict__ src = lower + ((size_t)row * 800 + r) * 32 + kseg;
        _Float16 ta[16];
        #pragma unroll
        for (int j = 0; j < 16; ++j) ta[j] = (_Float16)src[j];
        *(v8h*)(As + row * LDSTR + kseg)     = *(v8h*)&ta[0];
        *(v8h*)(As + row * LDSTR + kseg + 8) = *(v8h*)&ta[8];
    }
    if (tid < 64) {   // B: route_w[r][j][o][i] == [n][k], k contiguous
        int nrow = tid >> 1;
        int kseg = (tid & 1) * 16;
        const float* __restrict__ src = rw + (size_t)r * 1024 + nrow * 32 + kseg;
        _Float16 tb[16];
        #pragma unroll
        for (int j = 0; j < 16; ++j) tb[j] = (_Float16)src[j];
        *(v8h*)(Bs + nrow * LDSTR + kseg)     = *(v8h*)&tb[0];
        *(v8h*)(Bs + nrow * LDSTR + kseg + 8) = *(v8h*)&tb[8];
    }
    __syncthreads();

    const int wv   = tid >> 5;
    const int lane = tid & 31;
    const int g    = lane >> 4;
    const int lr   = lane & 15;
    const int mo   = wv * 16;

    v16h a;
    {
        int rowa = mo + lr;
        ((v8h*)&a)[0] = *(v8h*)(As + rowa * LDSTR + 8 * g);
        ((v8h*)&a)[1] = *(v8h*)(As + rowa * LDSTR + 16 + 8 * g);
    }
    v8f acc[2] = {};
    #pragma unroll
    for (int tn = 0; tn < 2; ++tn) {
        int rowb = tn * 16 + lr;
        v16h b;
        ((v8h*)&b)[0] = *(v8h*)(Bs + rowb * LDSTR + 8 * g);
        ((v8h*)&b)[1] = *(v8h*)(Bs + rowb * LDSTR + 16 + 8 * g);
        acc[tn] = __builtin_amdgcn_wmma_f32_16x16x32_f16(
            false, a, false, b, (short)0, acc[tn], false, false);
    }
    #pragma unroll
    for (int tn = 0; tn < 2; ++tn) {
        int n = tn * 16 + lr;
        int j = n >> 4, o = n & 15;
        #pragma unroll
        for (int rr = 0; rr < 8; ++rr) {
            int bidx = mo + 8 * g + rr;
            uhat[(((size_t)bidx * 800 + r) * 2 + j) * 16 + o] = acc[tn][rr];
        }
    }
}

// ---------------------------------------------------------------------------
// Kernel 5: dynamic routing (3 iters). One WG per batch element, logits in LDS.
// ---------------------------------------------------------------------------
__global__ __launch_bounds__(256) void routing_kernel(
    const float* __restrict__ uhat, float* __restrict__ vout)
{
    __shared__ float blog[1600];            // b[r][j]
    __shared__ float spart[8 * 32];
    __shared__ float vsh[32];

    const int b   = blockIdx.x;
    const int tid = threadIdx.x;
    const float* __restrict__ U = uhat + (size_t)b * 800 * 2 * 16;  // [r][j][o]

    for (int i = tid; i < 1600; i += 256) blog[i] = 0.f;
    __syncthreads();

    const int g  = tid >> 5;
    const int jo = tid & 31;
    const int j  = jo >> 4;
    const int o  = jo & 15;

    for (int it = 0; it < 3; ++it) {
        float part = 0.f;
        for (int rr = g * 100; rr < g * 100 + 100; ++rr) {
            float b0 = blog[rr * 2 + 0];
            float b1 = blog[rr * 2 + 1];
            float mx = fmaxf(b0, b1);
            float e0 = __expf(b0 - mx);
            float e1 = __expf(b1 - mx);
            float c  = (j == 0 ? e0 : e1) / (e0 + e1);
            part += c * U[(rr * 2 + j) * 16 + o];
        }
        spart[g * 32 + jo] = part;
        __syncthreads();

        if (tid < 32) {
            float s = 0.f;
            #pragma unroll
            for (int gg = 0; gg < 8; ++gg) s += spart[gg * 32 + tid];
            float n2 = s * s;                       // squash over o within 16 lanes
            n2 += __shfl_xor(n2, 8, 32);
            n2 += __shfl_xor(n2, 4, 32);
            n2 += __shfl_xor(n2, 2, 32);
            n2 += __shfl_xor(n2, 1, 32);
            float scale = (n2 / (1.f + n2)) / sqrtf(n2 + 1e-8f);
            vsh[tid] = s * scale;
        }
        __syncthreads();

        if (it < 2) {
            for (int idx = tid; idx < 1600; idx += 256) {
                int rr = idx >> 1, j2 = idx & 1;
                float dot = 0.f;
                #pragma unroll
                for (int oo = 0; oo < 16; ++oo)
                    dot += U[(rr * 2 + j2) * 16 + oo] * vsh[j2 * 16 + oo];
                blog[idx] += dot;
            }
            __syncthreads();
        }
    }
    if (tid < 32) vout[(size_t)b * 32 + tid] = vsh[tid];
}

// ---------------------------------------------------------------------------
extern "C" void kernel_launch(void* const* d_in, const int* in_sizes, int n_in,
                              void* d_out, int out_size, void* d_ws, size_t ws_size,
                              hipStream_t stream) {
    const float* feat  = (const float*)d_in[0];
    const float* w1    = (const float*)d_in[1];
    const float* b1    = (const float*)d_in[2];
    const float* gamma = (const float*)d_in[3];
    const float* beta  = (const float*)d_in[4];
    const float* mean  = (const float*)d_in[5];
    const float* var   = (const float*)d_in[6];
    const float* w2    = (const float*)d_in[7];
    const float* b2    = (const float*)d_in[8];
    const float* rw    = (const float*)d_in[9];
    float* out = (float*)d_out;

    // workspace layout (xpad is dead after conv1; p/uhat alias its region)
    char* ws = (char*)d_ws;
    _Float16* xpad = (_Float16*)ws;                        // 64*512*900 f16 = 58,982,400 B
    _Float16* x1   = (_Float16*)(ws + 58982400);           // 64*256*784 f16 = 25,690,112 B
    _Float16* w1t  = (_Float16*)(ws + 58982400 + 25690112);            //  2,359,296 B
    _Float16* w2t  = (_Float16*)(ws + 58982400 + 25690112 + 2359296);  // 10,616,832 B
    float* p       = (float*)ws;                           // aliases xpad: 6,553,600 B
    float* uhat    = (float*)(ws + 6553600);               // aliases xpad: 6,553,600 B

    float* lower = out;                                    // 64*800*32
    float* vout  = out + (size_t)64 * 800 * 32;            // 64*2*16

    pad_feat_kernel<<<115200, 256, 0, stream>>>(feat, xpad);
    w1_transform_kernel<<<4608, 256, 0, stream>>>(w1, w1t);
    w2_transform_kernel<<<20736, 256, 0, stream>>>(w2, w2t);
    conv1_bn_relu_kernel<<<784, 256, 0, stream>>>(xpad, w1t, b1, gamma, beta,
                                                  mean, var, x1);
    conv2_prim_kernel<<<100, 256, 0, stream>>>(x1, w2t, b2, p);
    squash_kernel<<<6400, 256, 0, stream>>>(p, lower);
    uhat_kernel<<<800, 128, 0, stream>>>(lower, rw, uhat);
    routing_kernel<<<64, 256, 0, stream>>>(uhat, vout);
}
